// ResidualGATLayer_1133871366389
// MI455X (gfx1250) — matrix operand order repaired and checked by reading, not compiled, not run on hardware
//
#include <hip/hip_runtime.h>
#include <math.h>

#define IN_C   256
#define HID    640          // 512 (4 heads x 128) + 128 (residual cols)
#define OUT_C  128
#define HEADS  4
#define LRELU  0.2f
#define LN_EPS 1e-5f

typedef __attribute__((ext_vector_type(16))) __bf16         v16bf;
typedef __attribute__((ext_vector_type(8)))  float          v8f;
typedef __attribute__((ext_vector_type(16))) unsigned short u16x16;
typedef __attribute__((ext_vector_type(8)))  unsigned short u16x8;

union BF16x16 { u16x16 u; v16bf b; };

__device__ __forceinline__ unsigned short f2bf(float f) {
  unsigned int u = __float_as_uint(f);
  u += 0x7FFFu + ((u >> 16) & 1u);          // round-to-nearest-even
  return (unsigned short)(u >> 16);
}

// monotonic float<->uint mapping so atomicMax(u32) orders floats correctly
__device__ __forceinline__ unsigned int fmap(float f) {
  unsigned int u = __float_as_uint(f);
  return (u & 0x80000000u) ? ~u : (u | 0x80000000u);
}
__device__ __forceinline__ float funmap(unsigned int u) {
  return __uint_as_float((u & 0x80000000u) ? (u & 0x7FFFFFFFu) : ~u);
}

// ---------------- prep: x -> bf16 ----------------
__global__ void k_convert_x(const float* __restrict__ x, unsigned short* __restrict__ xb, int n) {
  for (int t = blockIdx.x * blockDim.x + threadIdx.x; t < n; t += gridDim.x * blockDim.x)
    xb[t] = f2bf(x[t]);
}

// ---------------- prep: fused+swizzled weight ----------------
// wcs layout: [colTile ct 0..39][kstep kb 0..7][lane 0..31][e 0..15]
//   col = ct*16 + (lane&15), k = kb*32 + (e<8 ? (lane>>4)*8 + e : 16 + (lane>>4)*8 + e-8)      (ISA B 32x16 bf16 layout)
__global__ void k_prep_w(const float* __restrict__ W_lin, const float* __restrict__ W_res,
                         unsigned short* __restrict__ wcs) {
  int t = blockIdx.x * blockDim.x + threadIdx.x;     // 0 .. 256*640-1
  if (t >= IN_C * HID) return;
  int e    = t & 15;
  int lane = (t >> 4) & 31;
  int kb   = (t >> 9) & 7;
  int ct   = t >> 12;
  int col  = ct * 16 + (lane & 15);
  int k    = kb * 32 + ((e & 8) << 1) + ((lane >> 4) << 3) + (e & 7);
  float v  = (col < 512) ? W_lin[k * 512 + col] : W_res[k * OUT_C + (col - 512)];
  wcs[t] = f2bf(v);
}

// ---------------- GEMM: [N,256]bf16 x [256,640]bf16 -> [N,640]f32 via WMMA ----------------
// Block: 256 threads = 8 waves. One 16-row slab of A (16x256 bf16 = 8KB) staged in LDS once;
// each wave owns one 16-col tile of C and runs 8 ds_load+wmma steps with no further barriers.
__global__ __launch_bounds__(256) void k_gemm_wmma(
    const unsigned short* __restrict__ xb, const unsigned short* __restrict__ wcs,
    float* __restrict__ hres, int nrows) {
  __shared__ unsigned short sA[16 * IN_C];           // full 16x256 bf16 A slab
  const int lane = threadIdx.x & 31;
  const int wave = threadIdx.x >> 5;
  const int m0   = blockIdx.x * 16;
  const int ct   = blockIdx.y * 8 + wave;            // column tile 0..39
  const int half = lane >> 4;
  const int rowA = lane & 15;
  const int colC = (ct << 4) + (lane & 15);

  // cooperative A slab load: 512 chunks of 8 bf16 (16B); 2 chunks per thread
#pragma unroll
  for (int i = 0; i < 2; ++i) {
    int c   = threadIdx.x + i * 256;                 // 0..511
    int r   = c >> 5;                                // 0..15
    int off = (c & 31) * 8;                          // 0..248
    int row = m0 + r; if (row >= nrows) row = nrows - 1;
    *(u16x8*)(sA + r * IN_C + off) =
        *(const u16x8*)(xb + (size_t)row * IN_C + off);
  }
  __syncthreads();

  v8f acc = {};
  const unsigned short* sArow = sA + rowA * IN_C + half * 8;
#pragma unroll
  for (int kb = 0; kb < 8; ++kb) {
    const int k0 = kb << 5;
    // A (ISA 16-bit A 16x32): lane half selects K groups {half*8..} and {16+half*8..}
    BF16x16 A;
    {
      const u16x8 lo = *(const u16x8*)(sArow + k0);
      const u16x8 hi = *(const u16x8*)(sArow + k0 + 16);
#pragma unroll
      for (int i = 0; i < 8; ++i) { A.u[i] = lo[i]; A.u[8 + i] = hi[i]; }
    }
    // B: pre-swizzled, one contiguous 32B load per lane (immediate-offset b128 pair)
    BF16x16 B;
    B.u = *(const u16x16*)(wcs + ((((size_t)ct * 8 + kb) * 32 + lane) << 4));
    acc = __builtin_amdgcn_wmma_f32_16x16x32_bf16(false, A.b, false, B.b,
                                                  (short)0, acc, false, false);
  }
  // C/D layout: VGPR v -> row m0 + half*8 + v, col = lane&15
  const int rbase = m0 + (half << 3);
#pragma unroll
  for (int v = 0; v < 8; ++v) {
    int row = rbase + v;
    if (row < nrows) hres[(size_t)row * HID + colC] = acc[v];
  }
}

// ---------------- per-node attention logits + init of max/denom ----------------
__global__ void k_attn_logits(const float* __restrict__ hres,
                              const float* __restrict__ att_src, const float* __restrict__ att_dst,
                              float* __restrict__ a_src, float* __restrict__ a_dst,
                              unsigned int* __restrict__ mmax, float* __restrict__ denom, int n) {
  int t = blockIdx.x * blockDim.x + threadIdx.x;
  if (t >= n * HEADS) return;
  int node = t >> 2, hd = t & 3;
  const float* hp = hres + (size_t)node * HID + hd * OUT_C;
  const float* as = att_src + hd * OUT_C;
  const float* ad = att_dst + hd * OUT_C;
  float s0 = 0.f, s1 = 0.f;
  for (int c = 0; c < OUT_C; ++c) { float v = hp[c]; s0 += v * as[c]; s1 += v * ad[c]; }
  a_src[t] = s0; a_dst[t] = s1;
  mmax[t]  = fmap(-INFINITY);   // = 0x007FFFFF
  denom[t] = 0.f;
}

__device__ __forceinline__ void edge_endpoints(const int* ei, int E, int t, int& s, int& d) {
  if (t < E) { s = ei[t]; d = ei[E + t]; } else { s = d = t - E; }  // self-loops appended
}

// ---------------- edge pass 1: segment max ----------------
__global__ void k_edge_max(const int* __restrict__ ei, int E, int Etot,
                           const float* __restrict__ a_src, const float* __restrict__ a_dst,
                           unsigned int* __restrict__ mmax) {
  int t = blockIdx.x * blockDim.x + threadIdx.x;
  if (t >= Etot) return;
  int s, d; edge_endpoints(ei, E, t, s, d);
#pragma unroll
  for (int h = 0; h < HEADS; ++h) {
    float e = a_src[s * HEADS + h] + a_dst[d * HEADS + h];
    e = e > 0.f ? e : LRELU * e;
    atomicMax(&mmax[d * HEADS + h], fmap(e));
  }
}

// ---------------- edge pass 2: exp + segment sum ----------------
__global__ void k_edge_exp(const int* __restrict__ ei, int E, int Etot,
                           const float* __restrict__ a_src, const float* __restrict__ a_dst,
                           const unsigned int* __restrict__ mmax,
                           float* __restrict__ denom, float* __restrict__ exbuf) {
  int t = blockIdx.x * blockDim.x + threadIdx.x;
  if (t >= Etot) return;
  int s, d; edge_endpoints(ei, E, t, s, d);
#pragma unroll
  for (int h = 0; h < HEADS; ++h) {
    float e = a_src[s * HEADS + h] + a_dst[d * HEADS + h];
    e = e > 0.f ? e : LRELU * e;
    float ex = __expf(e - funmap(mmax[d * HEADS + h]));
    exbuf[(size_t)t * HEADS + h] = ex;
    atomicAdd(&denom[d * HEADS + h], ex);
  }
}

// ---------------- edge pass 3: aggregate, folding the head-mean (x0.25) ----------------
__global__ __launch_bounds__(256) void k_edge_aggregate(
    const int* __restrict__ ei, int E, int Etot,
    const float* __restrict__ hres, const float* __restrict__ exbuf,
    const float* __restrict__ denom, float* __restrict__ out_mean) {
  int wid = blockIdx.x * (blockDim.x >> 5) + (threadIdx.x >> 5);  // one wave32 per edge
  if (wid >= Etot) return;
  int lane = threadIdx.x & 31;
  int s, d; edge_endpoints(ei, E, wid, s, d);
  float al[HEADS];
#pragma unroll
  for (int h = 0; h < HEADS; ++h)
    al[h] = exbuf[(size_t)wid * HEADS + h] / (denom[d * HEADS + h] + 1e-16f);
  const int c0 = lane * 4;
  float4 acc = {0.f, 0.f, 0.f, 0.f};
#pragma unroll
  for (int h = 0; h < HEADS; ++h) {
    const float4 v = *(const float4*)(hres + (size_t)s * HID + h * OUT_C + c0);
    acc.x += al[h] * v.x; acc.y += al[h] * v.y;
    acc.z += al[h] * v.z; acc.w += al[h] * v.w;
  }
  float* o = out_mean + (size_t)d * OUT_C + c0;
  atomicAdd(o + 0, 0.25f * acc.x);
  atomicAdd(o + 1, 0.25f * acc.y);
  atomicAdd(o + 2, 0.25f * acc.z);
  atomicAdd(o + 3, 0.25f * acc.w);
}

// ---------------- final: bias + LayerNorm + residual ----------------
__global__ __launch_bounds__(256) void k_final_ln(
    const float* __restrict__ out_mean, const float* __restrict__ hres,
    const float* __restrict__ bias, const float* __restrict__ ln_w,
    const float* __restrict__ ln_b, const float* __restrict__ b_res,
    float* __restrict__ out, int n) {
  int node = blockIdx.x * (blockDim.x >> 5) + (threadIdx.x >> 5);  // one wave32 per node
  if (node >= n) return;
  int lane = threadIdx.x & 31;
  int c0 = lane * 4;
  float v[4];
  float s = 0.f;
#pragma unroll
  for (int j = 0; j < 4; ++j) {
    v[j] = out_mean[(size_t)node * OUT_C + c0 + j] + bias[c0 + j];
    s += v[j];
  }
#pragma unroll
  for (int m = 16; m >= 1; m >>= 1) s += __shfl_xor(s, m, 32);
  float mu = s * (1.f / OUT_C);
  float q = 0.f;
#pragma unroll
  for (int j = 0; j < 4; ++j) { float dv = v[j] - mu; q += dv * dv; }
#pragma unroll
  for (int m = 16; m >= 1; m >>= 1) q += __shfl_xor(q, m, 32);
  float inv = rsqrtf(q * (1.f / OUT_C) + LN_EPS);
#pragma unroll
  for (int j = 0; j < 4; ++j) {
    int c = c0 + j;
    float res = hres[(size_t)node * HID + 512 + c] + b_res[c];
    out[(size_t)node * OUT_C + c] = ln_w[c] * ((v[j] - mu) * inv) + ln_b[c] + res;
  }
}

extern "C" void kernel_launch(void* const* d_in, const int* in_sizes, int n_in,
                              void* d_out, int out_size, void* d_ws, size_t ws_size,
                              hipStream_t stream) {
  const float* x       = (const float*)d_in[0];
  const int*   ei      = (const int*)d_in[1];
  const float* W_lin   = (const float*)d_in[2];
  const float* att_src = (const float*)d_in[3];
  const float* att_dst = (const float*)d_in[4];
  const float* bias    = (const float*)d_in[5];
  const float* ln_w    = (const float*)d_in[6];
  const float* ln_b    = (const float*)d_in[7];
  const float* W_res   = (const float*)d_in[8];
  const float* b_res   = (const float*)d_in[9];

  const int N    = in_sizes[0] / IN_C;
  const int E    = in_sizes[1] / 2;
  const int Etot = E + N;            // self-loops appended

  char* ws = (char*)d_ws;
  size_t off = 0;
  auto take = [&](size_t bytes) -> char* {
    char* p = ws + off;
    off = (off + bytes + 255) & ~(size_t)255;
    return p;
  };
  unsigned short* xb      = (unsigned short*)take((size_t)N * IN_C * 2);
  unsigned short* wcs     = (unsigned short*)take((size_t)IN_C * HID * 2);
  float*          hres    = (float*)take((size_t)N * HID * 4);
  float*          a_src   = (float*)take((size_t)N * HEADS * 4);
  float*          a_dst   = (float*)take((size_t)N * HEADS * 4);
  unsigned int*   mmax    = (unsigned int*)take((size_t)N * HEADS * 4);
  float*          denom   = (float*)take((size_t)N * HEADS * 4);
  float*          exbuf   = (float*)take((size_t)Etot * HEADS * 4);
  float*          outmean = (float*)take((size_t)N * OUT_C * 4);
  (void)ws_size; (void)n_in; (void)out_size;

  // 1) convert x -> bf16; build fused+swizzled weight
  k_convert_x<<<dim3((N * IN_C + 255) / 256), dim3(256), 0, stream>>>(x, xb, N * IN_C);
  k_prep_w<<<dim3((IN_C * HID + 255) / 256), dim3(256), 0, stream>>>(W_lin, W_res, wcs);

  // 2) WMMA GEMM -> hres [N, 640] (h in cols 0..511, residual in 512..639)
  k_gemm_wmma<<<dim3((N + 15) / 16, HID / 128), dim3(256), 0, stream>>>(xb, wcs, hres, N);

  // 3) zero atomic accumulator (graph-capturable)
  hipMemsetAsync(outmean, 0, (size_t)N * OUT_C * 4, stream);

  // 4) per-node logits + init max/denom
  k_attn_logits<<<dim3((N * HEADS + 255) / 256), dim3(256), 0, stream>>>(
      hres, att_src, att_dst, a_src, a_dst, mmax, denom, N);

  // 5) segment softmax + aggregation over 850K edges
  k_edge_max<<<dim3((Etot + 255) / 256), dim3(256), 0, stream>>>(ei, E, Etot, a_src, a_dst, mmax);
  k_edge_exp<<<dim3((Etot + 255) / 256), dim3(256), 0, stream>>>(
      ei, E, Etot, a_src, a_dst, mmax, denom, exbuf);
  k_edge_aggregate<<<dim3((Etot + 7) / 8), dim3(256), 0, stream>>>(
      ei, E, Etot, hres, exbuf, denom, outmean);

  // 6) bias + LayerNorm + residual
  k_final_ln<<<dim3((N + 7) / 8), dim3(256), 0, stream>>>(
      outmean, hres, bias, ln_w, ln_b, b_res, (float*)d_out, N);
}
